// Decoupled_Rope_MHA_19018115187102
// MI455X (gfx1250) — matrix operand order repaired and checked by reading, not compiled, not run on hardware
//
#include <hip/hip_runtime.h>
#include <math.h>

// ---------- CDNA5 WMMA types ----------
typedef __attribute__((ext_vector_type(16))) __bf16 v16bf;
typedef __attribute__((ext_vector_type(8)))  __bf16 bh8;
typedef __attribute__((ext_vector_type(8)))  float  v8f;

union FragU { v16bf v; bh8 h[2]; };

#define WMMA_BF16(A, B, C) \
    __builtin_amdgcn_wmma_f32_16x16x32_bf16(false, (A), false, (B), (short)0, (C), false, false)

#define NEG_INF_F (-1.0e30f)

__device__ inline float half_max(float v) {
    v = fmaxf(v, __shfl_xor(v, 1));
    v = fmaxf(v, __shfl_xor(v, 2));
    v = fmaxf(v, __shfl_xor(v, 4));
    v = fmaxf(v, __shfl_xor(v, 8));
    return v;
}
__device__ inline float half_sum(float v) {
    v += __shfl_xor(v, 1);
    v += __shfl_xor(v, 2);
    v += __shfl_xor(v, 4);
    v += __shfl_xor(v, 8);
    return v;
}

__device__ inline bh8 pack_bf8(float4 a, float4 b) {
    bh8 r;
    r[0] = (__bf16)a.x; r[1] = (__bf16)a.y; r[2] = (__bf16)a.z; r[3] = (__bf16)a.w;
    r[4] = (__bf16)b.x; r[5] = (__bf16)b.y; r[6] = (__bf16)b.z; r[7] = (__bf16)b.w;
    return r;
}

// =====================================================================
// C[M,N] = A[M,K] @ B[N,K]^T   (A,B f32 row-major, staged to LDS as bf16)
// Requirements: M % 128 == 0, K % 32 == 0. N may be ragged (clamped
// staging rows; epilogue guards ncol < N).
// 128x128 block tile, K-step 32, double-buffered LDS (2-stage pipeline),
// 8 waves arranged 4(M) x 2(N), each wave -> 2x4 tiles of 16x16.
// =====================================================================
__global__ __launch_bounds__(256)
void gemm_bf16_wmma(const float* __restrict__ A, const float* __restrict__ Bm,
                    float* __restrict__ C, int M, int N, int K) {
    __shared__ __align__(16) __bf16 As[2 * 128 * 40];  // stride 40 bf16 = 80B
    __shared__ __align__(16) __bf16 Bs[2 * 128 * 40];

    const int tid  = threadIdx.x;
    const int lane = tid & 31;
    const int wave = tid >> 5;
    const int wm   = wave >> 1;      // 0..3
    const int wn   = wave & 1;       // 0..1
    const int m0   = blockIdx.y * 128;
    const int n0   = blockIdx.x * 128;

    v8f acc[2][4];
#pragma unroll
    for (int i = 0; i < 2; ++i)
#pragma unroll
        for (int j = 0; j < 4; ++j)
            acc[i][j] = (v8f){0.f, 0.f, 0.f, 0.f, 0.f, 0.f, 0.f, 0.f};

    // staging map: thread -> (row 0..127, 16-col half), branch-free
    const int srow = tid >> 1;          // 0..127
    const int scol = (tid & 1) * 16;    // 0 or 16
    const int brow = min(n0 + srow, N - 1);              // clamp ragged N
    const float* arow_p = A  + (size_t)(m0 + srow) * K + scol;
    const float* brow_p = Bm + (size_t)brow * K + scol;
    const int sidx = srow * 40 + scol;

    const int nk = K >> 5;

    // ---- prologue: stage tile 0 into buffer 0 ----
    {
        const float4* a4 = (const float4*)(arow_p);
        const float4* b4 = (const float4*)(brow_p);
        float4 a0 = a4[0], a1 = a4[1], a2 = a4[2], a3 = a4[3];
        float4 b0 = b4[0], b1 = b4[1], b2 = b4[2], b3 = b4[3];
        *(bh8*)&As[sidx]     = pack_bf8(a0, a1);
        *(bh8*)&As[sidx + 8] = pack_bf8(a2, a3);
        *(bh8*)&Bs[sidx]     = pack_bf8(b0, b1);
        *(bh8*)&Bs[sidx + 8] = pack_bf8(b2, b3);
    }

    const int base8 = (lane >> 4) << 3;
    const int kb16  = (lane >> 4) << 4;

    for (int it = 0; it < nk; ++it) {
        __syncthreads();
        const int cur = (it & 1) * (128 * 40);
        const bool more = (it + 1) < nk;   // block-uniform

        // kick next tile's global loads early (latency hidden behind WMMAs)
        float4 na0, na1, na2, na3, nb0, nb1, nb2, nb3;
        if (more) {
            const float4* a4 = (const float4*)(arow_p + (it + 1) * 32);
            const float4* b4 = (const float4*)(brow_p + (it + 1) * 32);
            na0 = a4[0]; na1 = a4[1]; na2 = a4[2]; na3 = a4[3];
            nb0 = b4[0]; nb1 = b4[1]; nb2 = b4[2]; nb3 = b4[3];
        }

        // A fragment: lane row = lane&15, contiguous 16 K at (lane>=16)*16
        v16bf af[2];
#pragma unroll
        for (int tm = 0; tm < 2; ++tm) {
            const __bf16* row = &As[cur + (wm * 32 + tm * 16 + (lane & 15)) * 40];
            FragU u;
            u.h[0] = *(const bh8*)(row + kb16);
            u.h[1] = *(const bh8*)(row + kb16 + 8);
            af[tm] = u.v;
        }
        // B fragment: lane col = lane&15, contiguous 16 K at (lane>=16)*16
        v16bf bf[4];
#pragma unroll
        for (int tn = 0; tn < 4; ++tn) {
            const __bf16* row = &Bs[cur + (wn * 64 + tn * 16 + (lane & 15)) * 40];
            FragU u;
            u.h[0] = *(const bh8*)(row + kb16);
            u.h[1] = *(const bh8*)(row + kb16 + 8);
            bf[tn] = u.v;
        }
#pragma unroll
        for (int tm = 0; tm < 2; ++tm)
#pragma unroll
            for (int tn = 0; tn < 4; ++tn)
                acc[tm][tn] = WMMA_BF16(af[tm], bf[tn], acc[tm][tn]);

        // stage next tile into the other buffer (disjoint from 'cur')
        if (more) {
            const int nxt = ((it + 1) & 1) * (128 * 40);
            *(bh8*)&As[nxt + sidx]     = pack_bf8(na0, na1);
            *(bh8*)&As[nxt + sidx + 8] = pack_bf8(na2, na3);
            *(bh8*)&Bs[nxt + sidx]     = pack_bf8(nb0, nb1);
            *(bh8*)&Bs[nxt + sidx + 8] = pack_bf8(nb2, nb3);
        }
    }

    // epilogue: C layout VGPR r -> M = r + (lane>=16)*8, N = lane&15
#pragma unroll
    for (int tm = 0; tm < 2; ++tm) {
        const int mbase = m0 + wm * 32 + tm * 16 + ((lane >> 4) << 3);
#pragma unroll
        for (int tn = 0; tn < 4; ++tn) {
            const int ncol = n0 + wn * 64 + tn * 16 + (lane & 15);
            if (ncol < N) {
#pragma unroll
                for (int r = 0; r < 8; ++r) {
                    C[(size_t)(mbase + r) * N + ncol] = acc[tm][tn][r];
                }
            }
        }
    }
}

// =====================================================================
// RoPE + head reshape. One block (256 thr) per token (b,s).
// qv:  [M,4096] f32 (Q | V),  kp: [M,1088] f32 (Kn heads | Kr)
// Writes bf16 q/k/v (B,H,S,128) for attention, f32 k_heads/v_heads to d_out.
// =====================================================================
__global__ __launch_bounds__(256)
void build_heads(const float* __restrict__ qv, const float* __restrict__ kp,
                 __bf16* __restrict__ qb, __bf16* __restrict__ kb, __bf16* __restrict__ vb,
                 float* __restrict__ k_out, float* __restrict__ v_out,
                 int Bn, int S, int H) {
    const int m = blockIdx.x;
    const int s = m % S;
    const int b = m / S;
    const int t = threadIdx.x;
    const float* qrow = qv + (size_t)m * 4096;
    const float* vrow = qrow + 2048;
    const float* krow = kp + (size_t)m * 1088;

    for (int idx = t; idx < 2048; idx += 256) {
        const int h = idx >> 7, j = idx & 127;
        // ---- Q ----
        float valq = qrow[idx];
        if (j >= 64) {
            const int jr = j - 64;
            const int fi = jr & 31;
            const float freq = __powf(10000.0f, -(float)fi * (1.0f / 64.0f));
            float sn, cs;
            __sincosf((float)s * freq, &sn, &cs);
            const float partner = (jr < 32) ? -qrow[(h << 7) + 64 + jr + 32]
                                            :  qrow[(h << 7) + 64 + jr - 32];
            valq = valq * cs + partner * sn;
        }
        const size_t o = (((size_t)(b * H + h)) * S + s) * 128 + j;
        qb[o] = (__bf16)valq;
        // ---- K ----
        float valk;
        if (j < 64) {
            valk = krow[h * 64 + j];
        } else {
            const int jr = j - 64;
            const int fi = jr & 31;
            const float freq = __powf(10000.0f, -(float)fi * (1.0f / 64.0f));
            float sn, cs;
            __sincosf((float)s * freq, &sn, &cs);
            const float kr = krow[1024 + jr];
            const float partner = (jr < 32) ? -krow[1024 + jr + 32] : krow[1024 + jr - 32];
            valk = kr * cs + partner * sn;
        }
        k_out[o] = valk;
        kb[o]    = (__bf16)valk;
        // ---- V ----
        const float valv = vrow[idx];
        v_out[o] = valv;
        vb[o]    = (__bf16)valv;
    }
}

// =====================================================================
// Causal flash attention, dh = 128. One block per (b, h, 128 q-rows).
// Each wave owns 16 q rows; 32-wide key tiles, double-buffered LDS.
// =====================================================================
__global__ __launch_bounds__(256)
void flash_attn(const __bf16* __restrict__ qb, const __bf16* __restrict__ kb,
                const __bf16* __restrict__ vb, float* __restrict__ attn_out,
                int Bn, int S, int H) {
    const int nqb    = S / 128;
    const int qb_idx = blockIdx.x % nqb;
    const int bh     = blockIdx.x / nqb;
    const int h      = bh % H;
    const int b      = bh / H;
    const int q0     = qb_idx * 128;

    const int tid   = threadIdx.x;
    const int lane  = tid & 31;
    const int wave  = tid >> 5;
    const int qbase = q0 + wave * 16;

    __shared__ __align__(16) __bf16 Ks[2 * 32 * 136];   // K tile [key][dh], padded
    __shared__ __align__(16) __bf16 Vt[2 * 128 * 40];   // V tile transposed [dh][key]
    __shared__ __align__(16) __bf16 Ps[8][16 * 40];     // per-wave P scratch [row][key]

    const __bf16* qptr = qb + ((size_t)(b * H + h) * S) * 128;
    const __bf16* kptr = kb + ((size_t)(b * H + h) * S) * 128;
    const __bf16* vptr = vb + ((size_t)(b * H + h) * S) * 128;

    // --- Q fragments (A-matrix 16x32 layout), dh=128 -> 4 K-steps ---
    const int base8 = (lane >> 4) << 3;
    const int kb16  = (lane >> 4) << 4;
    v16bf qf[4];
    {
        const __bf16* row = qptr + (size_t)(qbase + (lane & 15)) * 128;
#pragma unroll
        for (int ks = 0; ks < 4; ++ks) {
            FragU u;
            u.h[0] = *(const bh8*)(row + ks * 32 + kb16);
            u.h[1] = *(const bh8*)(row + ks * 32 + kb16 + 8);
            qf[ks] = u.v;
        }
    }

    float mrow[8], lrow[8];
    v8f o[8];
#pragma unroll
    for (int r = 0; r < 8; ++r) { mrow[r] = NEG_INF_F; lrow[r] = 0.0f; }
#pragma unroll
    for (int t8 = 0; t8 < 8; ++t8)
        o[t8] = (v8f){0.f, 0.f, 0.f, 0.f, 0.f, 0.f, 0.f, 0.f};

    const float scale = 0.08838834764831845f;  // 1/sqrt(128)
    const int ntile = (q0 + 128) >> 5;

    // staging map
    const int srow = tid >> 3;          // 0..31 (key within tile)
    const int scb  = (tid & 7) * 16;    // dh 0..112

    // ---- prologue: stage tile 0 into buffer 0 ----
    {
        const __bf16* src = kptr + (size_t)srow * 128 + scb;
        bh8 ka = *(const bh8*)src;
        bh8 kc = *(const bh8*)(src + 8);
        *(bh8*)&Ks[srow * 136 + scb]     = ka;
        *(bh8*)&Ks[srow * 136 + scb + 8] = kc;
        const __bf16* vs = vptr + (size_t)srow * 128 + scb;
        bh8 va = *(const bh8*)vs;
        bh8 vc = *(const bh8*)(vs + 8);
#pragma unroll
        for (int e = 0; e < 8; ++e) Vt[(scb + e) * 40 + srow]     = va[e];
#pragma unroll
        for (int e = 0; e < 8; ++e) Vt[(scb + 8 + e) * 40 + srow] = vc[e];
    }

    for (int it = 0; it < ntile; ++it) {
        __syncthreads();
        const int k0 = it * 32;
        const bool more = (it + 1) < ntile;  // block-uniform

        // kick next tile's global loads
        bh8 nka, nkc, nva, nvc;
        if (more) {
            const __bf16* src = kptr + (size_t)(k0 + 32 + srow) * 128 + scb;
            nka = *(const bh8*)src;
            nkc = *(const bh8*)(src + 8);
            const __bf16* vs = vptr + (size_t)(k0 + 32 + srow) * 128 + scb;
            nva = *(const bh8*)vs;
            nvc = *(const bh8*)(vs + 8);
        }

        if (k0 <= qbase + 15) {  // wave-uniform: EXEC stays all-ones for WMMA
            const __bf16* KsB = &Ks[(it & 1) * 32 * 136];
            const __bf16* VtB = &Vt[(it & 1) * 128 * 40];
            // ---- scores: Q(16x128) x K^T(128x32) -> two 16x16 tiles ----
            v8f sacc[2];
#pragma unroll
            for (int nt = 0; nt < 2; ++nt) {
                sacc[nt] = (v8f){0.f, 0.f, 0.f, 0.f, 0.f, 0.f, 0.f, 0.f};
                const int key = nt * 16 + (lane & 15);
#pragma unroll
                for (int ks = 0; ks < 4; ++ks) {
                    FragU u;
                    u.h[0] = *(const bh8*)&KsB[key * 136 + ks * 32 + kb16];
                    u.h[1] = *(const bh8*)&KsB[key * 136 + ks * 32 + kb16 + 8];
                    sacc[nt] = WMMA_BF16(qf[ks], u.v, sacc[nt]);
                }
            }
            // ---- online softmax per row ----
            __bf16* pw = &Ps[wave][0];
#pragma unroll
            for (int r = 0; r < 8; ++r) {
                const int mrow16 = r + ((lane >> 4) << 3);  // row within 16-tile
                const int qi = qbase + mrow16;
                float s0 = sacc[0][r] * scale;
                float s1 = sacc[1][r] * scale;
                const int ki0 = k0 + (lane & 15);
                if (ki0 > qi)      s0 = NEG_INF_F;
                if (ki0 + 16 > qi) s1 = NEG_INF_F;
                const float tmax = half_max(fmaxf(s0, s1));
                const float mnew = fmaxf(mrow[r], tmax);
                const float p0 = __expf(s0 - mnew);
                const float p1 = __expf(s1 - mnew);
                const float alpha = __expf(mrow[r] - mnew);
                lrow[r] = lrow[r] * alpha + half_sum(p0 + p1);
                mrow[r] = mnew;
#pragma unroll
                for (int t8 = 0; t8 < 8; ++t8) o[t8][r] *= alpha;
                pw[mrow16 * 40 + (lane & 15)]      = (__bf16)p0;
                pw[mrow16 * 40 + 16 + (lane & 15)] = (__bf16)p1;
            }
            asm volatile("s_wait_dscnt 0" ::: "memory");  // order P store -> P load (same wave)

            // ---- P (16x32) x V (32x128): 8 dh tiles ----
            FragU pu;
            const __bf16* prow = &Ps[wave][(lane & 15) * 40];
            pu.h[0] = *(const bh8*)(prow + kb16);
            pu.h[1] = *(const bh8*)(prow + kb16 + 8);
#pragma unroll
            for (int t8 = 0; t8 < 8; ++t8) {
                FragU vu;
                const __bf16* vr = &VtB[(t8 * 16 + (lane & 15)) * 40 + kb16];
                vu.h[0] = *(const bh8*)(vr);
                vu.h[1] = *(const bh8*)(vr + 8);
                o[t8] = WMMA_BF16(pu.v, vu.v, o[t8]);
            }
        }

        // stage next tile into the other buffer (disjoint from current)
        if (more) {
            const int nb = ((it + 1) & 1);
            __bf16* KsN = &Ks[nb * 32 * 136];
            __bf16* VtN = &Vt[nb * 128 * 40];
            *(bh8*)&KsN[srow * 136 + scb]     = nka;
            *(bh8*)&KsN[srow * 136 + scb + 8] = nkc;
#pragma unroll
            for (int e = 0; e < 8; ++e) VtN[(scb + e) * 40 + srow]     = nva[e];
#pragma unroll
            for (int e = 0; e < 8; ++e) VtN[(scb + 8 + e) * 40 + srow] = nvc[e];
        }
    }

    // ---- epilogue: attn_out layout [B, S, H*128] for the final GEMM ----
    const int D = H * 128;
#pragma unroll
    for (int t8 = 0; t8 < 8; ++t8) {
        const int col = h * 128 + t8 * 16 + (lane & 15);
#pragma unroll
        for (int r = 0; r < 8; ++r) {
            const int qi = qbase + r + ((lane >> 4) << 3);
            attn_out[((size_t)b * S + qi) * D + col] = o[t8][r] * (1.0f / lrow[r]);
        }
    }
}

// =====================================================================
extern "C" void kernel_launch(void* const* d_in, const int* in_sizes, int n_in,
                              void* d_out, int out_size, void* d_ws, size_t ws_size,
                              hipStream_t stream) {
    const float* x   = (const float*)d_in[0];
    const float* qkv = (const float*)d_in[1];
    const float* wk  = (const float*)d_in[2];
    const float* wo  = (const float*)d_in[3];

    const int Bn = 2, S = 2048, D = 2048, H = 16;
    const int M   = Bn * S;          // 4096
    const int NQV = 2 * D;           // 4096
    const int NK  = H * 64 + 64;     // 1088
    const size_t HM = (size_t)Bn * H * S * 128;  // head-major element count

    // workspace carve-up
    float*  qv   = (float*)d_ws;                       // M x 4096 f32
    float*  kp   = qv + (size_t)M * NQV;               // M x 1088 f32
    float*  attn = kp + (size_t)M * NK;                // M x 2048 f32
    __bf16* qbuf = (__bf16*)(attn + (size_t)M * D);    // (B,H,S,128) bf16
    __bf16* kbuf = qbuf + HM;
    __bf16* vbuf = kbuf + HM;

    float* out   = (float*)d_out;                      // (B,S,D)
    float* k_out = out + (size_t)M * D;                // (B,H,S,128)
    float* v_out = k_out + (size_t)M * D;              // (B,H,S,128)

    dim3 blk(256);
    // 1) QV = x @ qkv.T
    gemm_bf16_wmma<<<dim3(NQV / 128, M / 128), blk, 0, stream>>>(x, qkv, qv, M, NQV, D);
    // 2) K = x @ wk.T
    gemm_bf16_wmma<<<dim3((NK + 127) / 128, M / 128), blk, 0, stream>>>(x, wk, kp, M, NK, D);
    // 3) RoPE + head split; writes f32 k_heads/v_heads outputs + bf16 attention operands
    build_heads<<<dim3(M), blk, 0, stream>>>(qv, kp, qbuf, kbuf, vbuf, k_out, v_out, Bn, S, H);
    // 4) causal flash attention -> attn [B,S,D]
    flash_attn<<<dim3(Bn * H * (S / 128)), blk, 0, stream>>>(qbuf, kbuf, vbuf, attn, Bn, S, H);
    // 5) out = attn @ wo.T
    gemm_bf16_wmma<<<dim3(D / 128, M / 128), blk, 0, stream>>>(attn, wo, out, M, D, D);
}
